// GAT_layer_18502719111879
// MI455X (gfx1250) — matrix operand order repaired and checked by reading, not compiled, not run on hardware
//
#include <hip/hip_runtime.h>
#include <hip/hip_bf16.h>

// ---------------- problem constants ----------------
#define HID   512
#define NB    8
#define NN    1024
#define NHD   8
#define HU    64          // head dim
#define MTOT  (NB*NN)     // 8192 rows
#define ALPHA 0.1f
#define MASKV (-16384.0f)
#define LNEPS 1e-5f

typedef __attribute__((ext_vector_type(16))) __bf16 v16bf;
typedef __attribute__((ext_vector_type(8)))  __bf16 v8bf;
typedef __attribute__((ext_vector_type(8)))  float  v8f;
typedef __attribute__((ext_vector_type(4)))  unsigned int v4u;
typedef __attribute__((ext_vector_type(8)))  int    v8i;
typedef __attribute__((ext_vector_type(4)))  int    v4i;

static __device__ __forceinline__ v16bf join16(v8bf lo, v8bf hi) {
    union { v16bf w; v8bf h[2]; } u;
    u.h[0] = lo; u.h[1] = hi;
    return u.w;
}
static __device__ __forceinline__ v8f vzero8() {
    v8f z;
#pragma unroll
    for (int i = 0; i < 8; ++i) z[i] = 0.0f;
    return z;
}

// ---- CDNA5 async global->LDS copy (ASYNCcnt path, no VGPR round trip) ----
static __device__ __forceinline__ void async_g2l_b128(void* lds_dst, const void* gsrc) {
    uint32_t loff = (uint32_t)(uintptr_t)lds_dst;   // LDS byte offset (low 32 bits)
    asm volatile("global_load_async_to_lds_b128 %0, %1, off"
                 :: "v"(loff), "v"(gsrc)
                 : "memory");
}
template <int N>
static __device__ __forceinline__ void wait_async_le() {
    asm volatile("s_wait_asynccnt %0" :: "i"(N) : "memory");
}

// ---- CDNA5 Tensor Data Mover: 2-D bf16 tile load with LDS row padding ----
static __device__ __forceinline__ void tdm_load_2d(void* lds_dst, const void* gaddr,
                                                   uint32_t tensor_d0, uint32_t tensor_d1,
                                                   uint32_t stride_d0,
                                                   uint32_t tile_d0, uint32_t tile_d1,
                                                   uint32_t pad_iv_code, uint32_t pad_am_code) {
    uint64_t ga = (uint64_t)(uintptr_t)gaddr;
    v4u g0;
    g0[0] = 1u;                                                   // count=1, user D#
    g0[1] = (uint32_t)(uintptr_t)lds_dst;                         // lds_addr
    g0[2] = (uint32_t)ga;                                         // global_addr[31:0]
    g0[3] = (uint32_t)((ga >> 32) & 0x01FFFFFFu) | (2u << 30);    // ga[56:32] | type=2
    v8i g1;
    g1[0] = (int)((1u << 16)                  // data_size = 1 (2 bytes, bf16)
                | (1u << 20)                  // pad_enable
                | (pad_iv_code << 22)         // pad_interval
                | (pad_am_code << 25));       // pad_amount
    g1[1] = (int)((tensor_d0 & 0xFFFFu) << 16);
    g1[2] = (int)(((tensor_d0 >> 16) & 0xFFFFu) | ((tensor_d1 & 0xFFFFu) << 16));
    g1[3] = (int)(((tensor_d1 >> 16) & 0xFFFFu) | ((tile_d0 & 0xFFFFu) << 16));
    g1[4] = (int)(tile_d1 & 0xFFFFu);
    g1[5] = (int)stride_d0;
    g1[6] = 0;
    g1[7] = 0;
    v4i g2, g3;
#pragma unroll
    for (int i = 0; i < 4; ++i) { g2[i] = 0; g3[i] = 0; }
#if __clang_major__ >= 23
    v8i g4;
#pragma unroll
    for (int i = 0; i < 8; ++i) g4[i] = 0;
    __builtin_amdgcn_tensor_load_to_lds(g0, g1, g2, g3, g4, 0);
#else
    __builtin_amdgcn_tensor_load_to_lds(g0, g1, g2, g3, 0);
#endif
}

// ---------------- conversion / layout kernels ----------------
__global__ __launch_bounds__(256) void cvt_x_kernel(const float* __restrict__ x,
                                                    __bf16* __restrict__ xb) {
    int i = blockIdx.x * 256 + threadIdx.x;
    xb[i] = (__bf16)x[i];
}

__global__ __launch_bounds__(256) void cvt_wT_kernel(const float* __restrict__ W,
                                                     __bf16* __restrict__ Wt) {
    int idx = blockIdx.x * 256 + threadIdx.x;
    int k = idx >> 9, n = idx & 511;
    Wt[idx] = (__bf16)W[n * HID + k];
}

__global__ __launch_bounds__(256) void transpose_k_kernel(const __bf16* __restrict__ kb,
                                                          __bf16* __restrict__ kT) {
    int idx = blockIdx.x * 256 + threadIdx.x;
    int n  = idx & 1023;
    int d  = (idx >> 10) & 63;
    int bh = idx >> 16;
    int b = bh >> 3, h = bh & 7;
    kT[idx] = kb[(b * NN + n) * HID + h * HU + d];
}

__global__ __launch_bounds__(256) void reshuffle_v_kernel(const __bf16* __restrict__ vb,
                                                          __bf16* __restrict__ z0) {
    int idx = blockIdx.x * 256 + threadIdx.x;
    int d  = idx & 63;
    int n  = (idx >> 6) & 1023;
    int bh = idx >> 16;
    int b = bh >> 3, h = bh & 7;
    z0[idx] = vb[(b * NN + n) * HID + h * HU + d];
}

// ---------------- QKV projection GEMM: C = Xb * Wt + bias ----------------
// Double-buffered async-LDS pipeline: tile i+1 streams while tile i computes.
__global__ __launch_bounds__(256) void gemm_qkv_kernel(const __bf16* __restrict__ A,
                                                       const __bf16* __restrict__ Bt,
                                                       const float*  __restrict__ bias,
                                                       __bf16* __restrict__ C) {
    __shared__ __bf16 As[2][128 * 40];   // 80 B rows (16 DW data + 4 DW pad)
    __shared__ __bf16 Bs[2][32 * 136];

    const int tid = threadIdx.x;
    const int w = tid >> 5, l = tid & 31;
    const int hi = l >> 4, lm = l & 15;
    const int wm = (w >> 1) * 32;
    const int wn = (w & 1) * 64;
    const int mblk = blockIdx.y * 128;
    const int nblk = blockIdx.x * 128;

    // per-thread staging coordinates (2 A chunks + 2 B chunks = 4 asyncs/tile)
    const int ar = tid >> 2, ak8 = tid & 3;       // A chunks: rows ar, ar+64
    const int br = tid >> 4, bn8 = tid & 15;      // B chunks: rows br, br+16

    auto stage = [&](int k0, int buf) {
        async_g2l_b128(&As[buf][ar * 40 + ak8 * 8],
                       &A[(size_t)(mblk + ar) * HID + k0 + ak8 * 8]);
        async_g2l_b128(&As[buf][(ar + 64) * 40 + ak8 * 8],
                       &A[(size_t)(mblk + ar + 64) * HID + k0 + ak8 * 8]);
        async_g2l_b128(&Bs[buf][br * 136 + bn8 * 8],
                       &Bt[(size_t)(k0 + br) * HID + nblk + bn8 * 8]);
        async_g2l_b128(&Bs[buf][(br + 16) * 136 + bn8 * 8],
                       &Bt[(size_t)(k0 + br + 16) * HID + nblk + bn8 * 8]);
    };

    v8f acc[2][4];
#pragma unroll
    for (int i = 0; i < 2; ++i)
#pragma unroll
        for (int j = 0; j < 4; ++j) acc[i][j] = vzero8();

    stage(0, 0);                                   // prologue: tile 0 in flight
    for (int it = 0; it < HID / 32; ++it) {
        const int cur = it & 1;
        if (it + 1 < HID / 32) {
            stage((it + 1) * 32, cur ^ 1);         // keep next tile streaming
            wait_async_le<4>();                    // oldest 4 (tile it) complete
        } else {
            wait_async_le<0>();
        }
        __syncthreads();

        v16bf af[2], bfr[4];
#pragma unroll
        for (int i = 0; i < 2; ++i) {
            int m = wm + i * 16 + lm;
            af[i] = join16(*(const v8bf*)&As[cur][m * 40 + hi * 8],
                           *(const v8bf*)&As[cur][m * 40 + 16 + hi * 8]);
        }
        const int kk = lm + hi * 16;
#pragma unroll
        for (int j = 0; j < 4; ++j) {
            int n0 = wn + j * 16;
#pragma unroll
            for (int s = 0; s < 16; ++s)
                bfr[j][s] = Bs[cur][((s & 7) + hi * 8 + (s >> 3) * 16) * 136 + n0 + lm];
        }
#pragma unroll
        for (int i = 0; i < 2; ++i)
#pragma unroll
            for (int j = 0; j < 4; ++j)
                acc[i][j] = __builtin_amdgcn_wmma_f32_16x16x32_bf16(
                    false, af[i], false, bfr[j], (short)0, acc[i][j], false, false);
        __syncthreads();                            // reads done before buf reuse
    }

#pragma unroll
    for (int i = 0; i < 2; ++i)
#pragma unroll
        for (int j = 0; j < 4; ++j)
#pragma unroll
            for (int v = 0; v < 8; ++v) {
                int row = mblk + wm + i * 16 + hi * 8 + v;
                int col = nblk + wn + j * 16 + lm;
                C[(size_t)row * HID + col] = (__bf16)(acc[i][j][v] + bias[col]);
            }
}

// ---------------- fused scores + mask + softmax -> P (bf16) ----------------
__global__ __launch_bounds__(128) void scores_softmax_kernel(const __bf16* __restrict__ qb,
                                                             const __bf16* __restrict__ kT,
                                                             const int*    __restrict__ adj,
                                                             __bf16* __restrict__ P) {
    __shared__ __bf16 Qs[64 * 72];
    __shared__ __bf16 Kt[64 * 136];
    __shared__ __bf16 Sb[64 * 1024];
    __shared__ float  rmax[64], rsum[64];

    const int tid = threadIdx.x;
    const int w = tid >> 5, l = tid & 31;
    const int hi = l >> 4, lm = l & 15;
    const int rowblk = blockIdx.x * 64;
    const int h = blockIdx.y, b = blockIdx.z;
    const int bh = b * NHD + h;

    {   // Q tile 64x64: 4 async chunks per thread (rows qr, qr+16, qr+32, qr+48)
        const int qr = tid >> 3, qk8 = tid & 7;
#pragma unroll
        for (int s = 0; s < 4; ++s)
            async_g2l_b128(&Qs[(qr + s * 16) * 72 + qk8 * 8],
                           &qb[(size_t)(b * NN + rowblk + qr + s * 16) * HID + h * HU + qk8 * 8]);
    }
    wait_async_le<0>();
    __syncthreads();

    v16bf af0, af1;
    {
        int m = w * 16 + lm;
        af0 = join16(*(const v8bf*)&Qs[m * 72 + hi * 8],
                     *(const v8bf*)&Qs[m * 72 + 16 + hi * 8]);
        af1 = join16(*(const v8bf*)&Qs[m * 72 + 32 + hi * 8],
                     *(const v8bf*)&Qs[m * 72 + 48 + hi * 8]);
    }

    float mx[8], sm[8];
#pragma unroll
    for (int v = 0; v < 8; ++v) { mx[v] = -3.0e38f; sm[v] = 0.0f; }

    const int kr = tid >> 4, kn8 = tid & 15;      // K^T staging coords
    for (int cb = 0; cb < NN; cb += 128) {
#pragma unroll
        for (int s = 0; s < 8; ++s)               // K^T tile 64x128: 8 chunks/thread
            async_g2l_b128(&Kt[(kr + s * 8) * 136 + kn8 * 8],
                           &kT[(size_t)(bh * HU + kr + s * 8) * NN + cb + kn8 * 8]);
        if (cb + 128 < NN)
            __builtin_prefetch(&kT[(size_t)(bh * HU + (tid >> 1)) * NN + cb + 128], 0, 1);
        wait_async_le<0>();
        __syncthreads();

        const int kk = lm + hi * 16;
#pragma unroll
        for (int j = 0; j < 8; ++j) {
            int n0 = j * 16;
            v16bf b0, b1;
#pragma unroll
            for (int s = 0; s < 16; ++s) {
                int k = (s & 7) + hi * 8 + (s >> 3) * 16;
                b0[s] = Kt[k * 136 + n0 + lm];
                b1[s] = Kt[(k + 32) * 136 + n0 + lm];
            }
            v8f acc = vzero8();
            acc = __builtin_amdgcn_wmma_f32_16x16x32_bf16(false, af0, false, b0,
                                                          (short)0, acc, false, false);
            acc = __builtin_amdgcn_wmma_f32_16x16x32_bf16(false, af1, false, b1,
                                                          (short)0, acc, false, false);
            int col = cb + n0 + lm;
#pragma unroll
            for (int v = 0; v < 8; ++v) {
                int rloc = w * 16 + hi * 8 + v;
                int rowg = rowblk + rloc;
                float s = acc[v];
                if (adj[(size_t)(b * NN + rowg) * NN + col] == 0) s = MASKV;
                Sb[rloc * NN + col] = (__bf16)s;
                float tm = s;
#pragma unroll
                for (int d = 8; d >= 1; d >>= 1) tm = fmaxf(tm, __shfl_xor(tm, d, 32));
                float nm = fmaxf(mx[v], tm);
                float e = __expf(s - nm);
                float ts = e;
#pragma unroll
                for (int d = 8; d >= 1; d >>= 1) ts += __shfl_xor(ts, d, 32);
                sm[v] = sm[v] * __expf(mx[v] - nm) + ts;
                mx[v] = nm;
            }
        }
        __syncthreads();
    }

    if (lm == 0) {
#pragma unroll
        for (int v = 0; v < 8; ++v) {
            int rloc = w * 16 + hi * 8 + v;
            rmax[rloc] = mx[v];
            rsum[rloc] = sm[v];
        }
    }
    __syncthreads();

    for (int idx = tid; idx < 64 * 1024; idx += 128) {
        int r = idx >> 10, c = idx & 1023;
        float p = __expf((float)Sb[idx] - rmax[r]) / rsum[r];
        P[((size_t)bh * NN + rowblk + r) * NN + c] = (__bf16)p;
    }
}

// ---------------- hop GEMM: zout = (1-a) * P*zin + a * zinit ----------------
// Double-buffered TDM pipeline: tensor_load_to_lds for tile i+1 streams while
// WMMAs consume tile i (wave0 drives P tiles, wave1 drives Z tiles).
__global__ __launch_bounds__(256) void hop_kernel(const __bf16* __restrict__ P,
                                                  const __bf16* __restrict__ zin,
                                                  const __bf16* __restrict__ zinit,
                                                  __bf16* __restrict__ zout,
                                                  float*  __restrict__ hout,
                                                  int write_h) {
    __shared__ __bf16 As[2][128 * 40];   // 80 B rows: 16 DW data + 4 DW pad (TDM)
    __shared__ __bf16 Bs[2][32 * 72];    // 144 B rows: 32 DW data + 4 DW pad (TDM)

    const int tid = threadIdx.x;
    const int w = tid >> 5, l = tid & 31;
    const int hi = l >> 4, lm = l & 15;
    const int mblk = blockIdx.x * 128;
    const int h = blockIdx.y, b = blockIdx.z;
    const int bh = b * NHD + h;
    const __bf16* Pb = P + (size_t)bh * NN * NN;
    const __bf16* Zb = zin + (size_t)bh * NN * HU;

    v8f acc[4];
#pragma unroll
    for (int j = 0; j < 4; ++j) acc[j] = vzero8();

    // prologue: tile 0 in flight
    if (w == 0) {
        tdm_load_2d(&As[0][0], Pb + (size_t)mblk * NN,
                    NN, NN, NN, 32, 128, /*pad 16DW+4DW*/3, 3);
    } else if (w == 1) {
        tdm_load_2d(&Bs[0][0], Zb,
                    HU, NN, HU, 64, 32, /*pad 32DW+4DW*/4, 3);
    }

    for (int it = 0; it < NN / 32; ++it) {
        const int cur = it & 1;
        const int k0n = (it + 1) * 32;
        if (w == 0) {
            if (it + 1 < NN / 32) {
                tdm_load_2d(&As[cur ^ 1][0], Pb + (size_t)mblk * NN + k0n,
                            NN, NN, NN, 32, 128, 3, 3);
                __builtin_amdgcn_s_wait_tensorcnt((short)1);   // tile it done
            } else {
                __builtin_amdgcn_s_wait_tensorcnt((short)0);
            }
        } else if (w == 1) {
            if (it + 1 < NN / 32) {
                tdm_load_2d(&Bs[cur ^ 1][0], Zb + (size_t)k0n * HU,
                            HU, NN, HU, 64, 32, 4, 3);
                __builtin_amdgcn_s_wait_tensorcnt((short)1);
            } else {
                __builtin_amdgcn_s_wait_tensorcnt((short)0);
            }
        }
        __syncthreads();

        int m = w * 16 + lm;
        v16bf af = join16(*(const v8bf*)&As[cur][m * 40 + hi * 8],
                          *(const v8bf*)&As[cur][m * 40 + 16 + hi * 8]);
        const int kk = lm + hi * 16;
#pragma unroll
        for (int j = 0; j < 4; ++j) {
            v16bf bfr;
#pragma unroll
            for (int s = 0; s < 16; ++s)
                bfr[s] = Bs[cur][((s & 7) + hi * 8 + (s >> 3) * 16) * 72 + j * 16 + lm];
            acc[j] = __builtin_amdgcn_wmma_f32_16x16x32_bf16(
                false, af, false, bfr, (short)0, acc[j], false, false);
        }
        __syncthreads();                            // reads done before buf reuse
    }

#pragma unroll
    for (int j = 0; j < 4; ++j)
#pragma unroll
        for (int v = 0; v < 8; ++v) {
            int row = mblk + w * 16 + hi * 8 + v;
            int col = j * 16 + lm;
            size_t zidx = ((size_t)bh * NN + row) * HU + col;
            float r = (1.0f - ALPHA) * acc[j][v] + ALPHA * (float)zinit[zidx];
            zout[zidx] = (__bf16)r;
            if (write_h)
                hout[(size_t)(b * NN + row) * HID + h * HU + col] = r;
        }
}

// ---------------- residual + LayerNorm ----------------
__global__ __launch_bounds__(256) void resid_ln_kernel(const float* __restrict__ x,
                                                       const float* __restrict__ hp,
                                                       const float* __restrict__ gamma,
                                                       const float* __restrict__ beta,
                                                       float* __restrict__ out) {
    __shared__ float red[8];
    const int m = blockIdx.x;
    const int tid = threadIdx.x;
    float y0 = x[(size_t)m * HID + tid] + hp[(size_t)m * HID + tid];
    float y1 = x[(size_t)m * HID + tid + 256] + hp[(size_t)m * HID + tid + 256];

    float s = y0 + y1;
#pragma unroll
    for (int d = 16; d >= 1; d >>= 1) s += __shfl_xor(s, d, 32);
    if ((tid & 31) == 0) red[tid >> 5] = s;
    __syncthreads();
    float mu = 0.0f;
#pragma unroll
    for (int i = 0; i < 8; ++i) mu += red[i];
    mu *= (1.0f / HID);
    __syncthreads();

    float d0 = y0 - mu, d1 = y1 - mu;
    s = d0 * d0 + d1 * d1;
#pragma unroll
    for (int d = 16; d >= 1; d >>= 1) s += __shfl_xor(s, d, 32);
    if ((tid & 31) == 0) red[tid >> 5] = s;
    __syncthreads();
    float var = 0.0f;
#pragma unroll
    for (int i = 0; i < 8; ++i) var += red[i];
    var *= (1.0f / HID);

    float rs = rsqrtf(var + LNEPS);
    out[(size_t)m * HID + tid]       = d0 * rs * gamma[tid] + beta[tid];
    out[(size_t)m * HID + tid + 256] = d1 * rs * gamma[tid + 256] + beta[tid + 256];
}

// ---------------- launch ----------------
extern "C" void kernel_launch(void* const* d_in, const int* in_sizes, int n_in,
                              void* d_out, int out_size, void* d_ws, size_t ws_size,
                              hipStream_t stream) {
    (void)in_sizes; (void)n_in; (void)out_size; (void)ws_size;
    const float* x     = (const float*)d_in[0];
    const int*   adj   = (const int*)d_in[1];
    const float* Wq    = (const float*)d_in[2];
    const float* bq    = (const float*)d_in[3];
    const float* Wk    = (const float*)d_in[4];
    const float* bk    = (const float*)d_in[5];
    const float* Wv    = (const float*)d_in[6];
    const float* bv    = (const float*)d_in[7];
    const float* gamma = (const float*)d_in[8];
    const float* beta  = (const float*)d_in[9];
    float* out = (float*)d_out;

    char* ws = (char*)d_ws;
    size_t off = 0;
    auto take = [&](size_t bytes) { char* p = ws + off; off += (bytes + 255) & ~(size_t)255; return p; };
    __bf16* xb  = (__bf16*)take((size_t)MTOT * HID * 2);
    __bf16* wtq = (__bf16*)take((size_t)HID * HID * 2);
    __bf16* wtk = (__bf16*)take((size_t)HID * HID * 2);
    __bf16* wtv = (__bf16*)take((size_t)HID * HID * 2);
    __bf16* qb  = (__bf16*)take((size_t)MTOT * HID * 2);
    __bf16* kb  = (__bf16*)take((size_t)MTOT * HID * 2);
    __bf16* vb  = (__bf16*)take((size_t)MTOT * HID * 2);
    __bf16* kT  = (__bf16*)take((size_t)NB * NHD * HU * NN * 2);
    __bf16* z0  = (__bf16*)take((size_t)NB * NHD * NN * HU * 2);
    __bf16* z1  = (__bf16*)take((size_t)NB * NHD * NN * HU * 2);
    __bf16* z2  = (__bf16*)take((size_t)NB * NHD * NN * HU * 2);
    __bf16* Pm  = (__bf16*)take((size_t)NB * NHD * NN * NN * 2);
    float*  hpr = (float*) take((size_t)MTOT * HID * 4);

    cvt_x_kernel<<<(MTOT * HID) / 256, 256, 0, stream>>>(x, xb);
    cvt_wT_kernel<<<(HID * HID) / 256, 256, 0, stream>>>(Wq, wtq);
    cvt_wT_kernel<<<(HID * HID) / 256, 256, 0, stream>>>(Wk, wtk);
    cvt_wT_kernel<<<(HID * HID) / 256, 256, 0, stream>>>(Wv, wtv);

    dim3 ggrid(HID / 128, MTOT / 128);
    gemm_qkv_kernel<<<ggrid, 256, 0, stream>>>(xb, wtq, bq, qb);
    gemm_qkv_kernel<<<ggrid, 256, 0, stream>>>(xb, wtk, bk, kb);
    gemm_qkv_kernel<<<ggrid, 256, 0, stream>>>(xb, wtv, bv, vb);

    transpose_k_kernel<<<(NB * NHD * HU * NN) / 256, 256, 0, stream>>>(kb, kT);
    reshuffle_v_kernel<<<(NB * NHD * NN * HU) / 256, 256, 0, stream>>>(vb, z0);

    scores_softmax_kernel<<<dim3(NN / 64, NHD, NB), 128, 0, stream>>>(qb, kT, adj, Pm);

    dim3 hgrid(NN / 128, NHD, NB);
    hop_kernel<<<hgrid, 256, 0, stream>>>(Pm, z0, z0, z1, hpr, 0);
    hop_kernel<<<hgrid, 256, 0, stream>>>(Pm, z1, z0, z2, hpr, 0);
    hop_kernel<<<hgrid, 256, 0, stream>>>(Pm, z2, z0, z1, hpr, 0);
    hop_kernel<<<hgrid, 256, 0, stream>>>(Pm, z1, z0, z2, hpr, 1);

    resid_ln_kernel<<<MTOT, 256, 0, stream>>>(x, hpr, gamma, beta, out);
}
